// GenNet_8504035246156
// MI455X (gfx1250) — matrix operand order repaired and checked by reading, not compiled, not run on hardware
//
#include <hip/hip_runtime.h>

// ---------------------------------------------------------------------------
// GenNet on MI455X (gfx1250): fc1 -> 128x (LSTM0 -> LSTM1, zero state) -> fc2
// f16 WMMA (v_wmma_f32_16x16x32_f16) with f32 accumulate. f-gate skipped
// (c_prev == 0). Whole recurrence fused in one kernel; x/h tiles in LDS.
// Final: single base pointer per stream with all gate/layer/tile/row
// displacements folded into global/ds instruction immediates; (256,4)
// register budget -- keeps allocation <=256 VGPRs so the hot loop carries
// zero s_set_vgpr_msb bank-switch overhead (confirmed in round-4 disasm).
// ---------------------------------------------------------------------------

typedef _Float16 half16 __attribute__((ext_vector_type(16)));
typedef _Float16 half8  __attribute__((ext_vector_type(8)));
typedef float    v8f    __attribute__((ext_vector_type(8)));

#define HID    256
#define GATES  1024
#define BATCH  2048
#define TSTEPS 128
#define EMB    300
#define EMBP   304         // fc2 weight rows padded to 19 full 16-col tiles
#define NTILE2 (EMBP / 16) // 19
#define LDSW   264         // 256 + 8 halves pad -> conflict-free b128 loads
#define NW     (GATES * HID)      // elements per LSTM weight matrix (262144)
#define NFC1   (HID * 32)         // 8192
#define NFC2   (EMBP * HID)       // 77824 (padded)

__device__ __forceinline__ v8f wmma_f16(half16 a, half16 b, v8f c) {
    return __builtin_amdgcn_wmma_f32_16x16x32_f16(false, a, false, b,
                                                  (short)0, c, false, false);
}

__device__ __forceinline__ float fsigmoid(float x) {
    float e = __expf(-x);
    return __builtin_amdgcn_rcpf(1.0f + e);
}
__device__ __forceinline__ float ftanh(float x) {
    float e = __expf(-2.0f * x);
    return (1.0f - e) * __builtin_amdgcn_rcpf(1.0f + e);
}

// A-fragment (16x32 f16) from an LDS row pointer (&buf[m][hi*16]):
// lane m (hi=0): K 0..15 ; lane 16+m (hi=1): K 16..31
// k0 displacement folds into the ds_load immediate.
__device__ __forceinline__ half16 load_a(const _Float16* arow, int k0) {
    union { half16 v; half8 h[2]; } u;
    u.h[0] = *(const half8*)(arow + k0);
    u.h[1] = *(const half8*)(arow + k0 + 8);
    return u.v;
}

__global__ __launch_bounds__(256, 4) void gennet_main(
        const float* __restrict__ tag,   const float* __restrict__ fc1_b,
        const float* __restrict__ b_ih0, const float* __restrict__ b_hh0,
        const float* __restrict__ b_ih1, const float* __restrict__ b_hh1,
        const float* __restrict__ fc2_b,
        const _Float16* __restrict__ fc1h,   // [256][32]
        const _Float16* __restrict__ w0h,    // [1024][256]; w1h == w0h + NW
        const _Float16* __restrict__ fc2h,   // [304][256] (rows >=300 zero)
        float* __restrict__ out) {
    __shared__ alignas(32) _Float16 xbuf[16][LDSW];
    __shared__ alignas(32) _Float16 hbuf[16][LDSW];

    const int b0   = blockIdx.x * 16;
    const int tid  = threadIdx.x;
    const int wave = tid >> 5;
    const int lane = tid & 31;
    const int m    = lane & 15;
    const int hi   = lane >> 4;

    // ---- time-invariant per-lane bases (everything else is an immediate) --
    const _Float16* xrow = &xbuf[m][hi * 16];    // A-fragment row base
    const _Float16* hrow = &hbuf[m][hi * 16];
    _Float16* xst = &xbuf[hi * 8][wave * 32 + m]; // h-store base (+tt*16+r*LDSW)
    _Float16* hst = &hbuf[hi * 8][wave * 32 + m];
    // one base for ALL 96 cell weight loads: + L*NW + gate/tt/kt consts
    const _Float16* pw = w0h + (unsigned)(wave * 32 + m) * HID + hi * 16;
    // one base for all fc2 weight loads: + j*128*HID + kt*32
    const _Float16* pf = fc2h + (unsigned)(wave * 16 + m) * HID + hi * 16;
    // one running output pointer: + j*128 + r*TSTEPS*EMB, advanced EMB per t
    float* po = out + (size_t)(b0 + hi * 8) * TSTEPS * EMB + (wave * 16 + m);

    // combined biases (b_ih + b_hh), hoisted
    float cbi[2][2], cbg[2][2], cbo[2][2];
#pragma unroll
    for (int tt = 0; tt < 2; ++tt) {
        const int n = wave * 32 + tt * 16 + m;
        cbi[0][tt] = b_ih0[n]       + b_hh0[n];
        cbg[0][tt] = b_ih0[n + 512] + b_hh0[n + 512];
        cbo[0][tt] = b_ih0[n + 768] + b_hh0[n + 768];
        cbi[1][tt] = b_ih1[n]       + b_hh1[n];
        cbg[1][tt] = b_ih1[n + 512] + b_hh1[n + 512];
        cbo[1][tt] = b_ih1[n + 768] + b_hh1[n + 768];
    }

    // fc2 tile ownership: waves 0..2 own 3 tiles, waves 3..7 own 2.
    const int nt2 = (wave < NTILE2 - 16) ? 3 : 2;
    float fbias[3];
    bool  ok[3];
#pragma unroll
    for (int j = 0; j < 3; ++j) {
        const int n = wave * 16 + m + 128 * j;   // may reach 303 (padded)
        ok[j]    = (j < nt2) && (n < EMB);
        fbias[j] = ok[j] ? fc2_b[n] : 0.0f;
    }

    // ---- cell: gates from `arow` LDS tile, h -> `st` LDS base. L constant --
    auto cell = [&](const _Float16* arow, _Float16* st, const unsigned LOFF,
                    const float* bi, const float* bg, const float* bo) {
        v8f aI[2], aG[2], aO[2];
#pragma unroll
        for (int tt = 0; tt < 2; ++tt) { aI[tt] = {}; aG[tt] = {}; aO[tt] = {}; }
#pragma unroll
        for (int kt = 0; kt < 8; ++kt) {
            const int k0 = kt * 32;
            half16 a = load_a(arow, k0);
#pragma unroll
            for (int tt = 0; tt < 2; ++tt) {
                const unsigned od = LOFF + tt * (16u * HID) + k0;
                half16 bI = *(const half16*)(pw + od);                 // i gate
                half16 bG = *(const half16*)(pw + od + 512u * HID);    // g gate
                half16 bO = *(const half16*)(pw + od + 768u * HID);    // o gate
                aI[tt] = wmma_f16(a, bI, aI[tt]);
                aG[tt] = wmma_f16(a, bG, aG[tt]);
                aO[tt] = wmma_f16(a, bO, aO[tt]);
            }
        }
#pragma unroll
        for (int tt = 0; tt < 2; ++tt) {
#pragma unroll
            for (int r = 0; r < 8; ++r) {
                float iv = aI[tt][r] + bi[tt];
                float gv = aG[tt][r] + bg[tt];
                float ov = aO[tt][r] + bo[tt];
                float c  = fsigmoid(iv) * ftanh(gv);
                float h  = fsigmoid(ov) * ftanh(c);
                st[tt * 16 + r * LDSW] = (_Float16)h;  // all in ds imm
            }
        }
    };

    // ---- fc1: x0 = tag @ fc1_w^T + b (K = 32 -> one WMMA per tile) ----
    {
        const float* trow = tag + (size_t)(b0 + m) * 32;
        half16 a;
#pragma unroll
        for (int j = 0; j < 8; ++j) {
            a[j]     = (_Float16)trow[hi * 16 + j];
            a[8 + j] = (_Float16)trow[hi * 16 + 8 + j];
        }
#pragma unroll
        for (int tt = 0; tt < 2; ++tt) {
            const int n = wave * 32 + tt * 16 + m;
            half16 b = *(const half16*)(fc1h + (unsigned)n * 32 + hi * 16);
            v8f c = {};
            c = wmma_f16(a, b, c);
            const float bv = fc1_b[n];
#pragma unroll
            for (int r = 0; r < 8; ++r)
                xst[tt * 16 + r * LDSW] = (_Float16)(c[r] + bv);
        }
    }
    __syncthreads();

    // ---- 128 fused timesteps: cell0 -> cell1 -> fc2 ----
#pragma unroll 1
    for (int t = 0; t < TSTEPS; ++t) {
        cell(xrow, hst, 0u,  cbi[0], cbg[0], cbo[0]);
        __syncthreads();                   // hbuf ready; fc2(t-1) reads done
        cell(hrow, xst, (unsigned)NW, cbi[1], cbg[1], cbo[1]);
        __syncthreads();                   // xbuf ready for fc2 + next cell0

        // fc2: kt-outer so only one A-fragment + <=3 accumulators stay live
        v8f acc[3];
#pragma unroll
        for (int j = 0; j < 3; ++j) acc[j] = {};
#pragma unroll
        for (int kt = 0; kt < 8; ++kt) {
            const int k0 = kt * 32;
            half16 a = load_a(xrow, k0);
#pragma unroll
            for (int j = 0; j < 3; ++j) {
                if (j < nt2) {             // wave-uniform guard
                    half16 b = *(const half16*)(pf + j * (128u * HID) + k0);
                    acc[j] = wmma_f16(a, b, acc[j]);
                }
            }
        }
#pragma unroll
        for (int j = 0; j < 3; ++j) {
            if (ok[j]) {
#pragma unroll
                for (int r = 0; r < 8; ++r)
                    po[j * 128 + (size_t)r * TSTEPS * EMB] = acc[j][r] + fbias[j];
            }
        }
        po += EMB;                         // advance one timestep
    }
}

// ---- one-shot f32 -> f16 weight conversion into workspace ----
__global__ void convert_weights(const float* __restrict__ w0,
                                const float* __restrict__ w1,
                                const float* __restrict__ fc1w,
                                const float* __restrict__ fc2w,
                                _Float16* __restrict__ w0h,
                                _Float16* __restrict__ w1h,
                                _Float16* __restrict__ fc1h,
                                _Float16* __restrict__ fc2h) {
    const int total = 2 * NW + NFC1 + NFC2;
    for (int i = blockIdx.x * blockDim.x + threadIdx.x; i < total;
         i += gridDim.x * blockDim.x) {
        if (i < NW) {
            w0h[i] = (_Float16)w0[i];
        } else if (i < 2 * NW) {
            w1h[i - NW] = (_Float16)w1[i - NW];
        } else if (i < 2 * NW + NFC1) {
            fc1h[i - 2 * NW] = (_Float16)fc1w[i - 2 * NW];
        } else {
            const int j = i - 2 * NW - NFC1;
            const int row = j / HID, col = j % HID;
            fc2h[j] = (row < EMB) ? (_Float16)fc2w[(size_t)row * HID + col]
                                  : (_Float16)0.0f;
        }
    }
}

extern "C" void kernel_launch(void* const* d_in, const int* in_sizes, int n_in,
                              void* d_out, int out_size, void* d_ws, size_t ws_size,
                              hipStream_t stream) {
    (void)in_sizes; (void)n_in; (void)out_size; (void)ws_size;
    const float* tag   = (const float*)d_in[0];
    const float* fc1w  = (const float*)d_in[1];
    const float* fc1b  = (const float*)d_in[2];
    const float* w_ih0 = (const float*)d_in[3];
    // d_in[4] = w_hh0 unused (zero state)
    const float* b_ih0 = (const float*)d_in[5];
    const float* b_hh0 = (const float*)d_in[6];
    const float* w_ih1 = (const float*)d_in[7];
    // d_in[8] = w_hh1 unused
    const float* b_ih1 = (const float*)d_in[9];
    const float* b_hh1 = (const float*)d_in[10];
    const float* fc2w  = (const float*)d_in[11];
    const float* fc2b  = (const float*)d_in[12];
    // d_in[13] = seq_len (== 128, compile-time constant here)

    // Workspace layout (contiguous -- gennet_main relies on w1h == w0h + NW):
    _Float16* w0h  = (_Float16*)d_ws;
    _Float16* w1h  = w0h  + NW;
    _Float16* fc1h = w1h  + NW;
    _Float16* fc2h = fc1h + NFC1;
    float* out = (float*)d_out;

    const int total = 2 * NW + NFC1 + NFC2;
    convert_weights<<<(total + 255) / 256, 256, 0, stream>>>(
        w_ih0, w_ih1, fc1w, fc2w, w0h, w1h, fc1h, fc2h);

    gennet_main<<<BATCH / 16, 256, 0, stream>>>(
        tag, fc1b, b_ih0, b_hh0, b_ih1, b_hh1, fc2b,
        fc1h, w0h, fc2h, out);
}
